// LIFModel_9775345566183
// MI455X (gfx1250) — matrix operand order repaired and checked by reading, not compiled, not run on hardware
//
#include <hip/hip_runtime.h>
#include <math.h>

// ---------------------------------------------------------------------------
// Recurrent LIF SNN on gfx1250: persistent grid, bf16 WMMA GEMMs, f32 accum.
// ---------------------------------------------------------------------------

typedef __attribute__((ext_vector_type(16))) __bf16 v16bf;
typedef __attribute__((ext_vector_type(8)))  __bf16 v8bf;
typedef __attribute__((ext_vector_type(8)))  float  v8f;

#define T_STEPS 1000
#define B_SZ    64
#define NI      700
#define NI_PAD  704          // 22 K-tiles of 32
#define NF      1024         // 32 K-tiles of 32 / 64 N-tiles of 16
#define NO      20
#define NO_PAD  32

#define NWG_HID 16
#define NWG_TOT 17           // + 1 readout block
#define THREADS 256          // 8 wave32s

// ---- workspace byte layout (all offsets 1KB aligned) ----------------------
#define BAR_OFF    0
#define XBUF_OFF   1024
#define XBUF_ELEMS (B_SZ * NI_PAD)                        // bf16 elems / buffer
#define ZBUF_OFF   (XBUF_OFF + 2 * XBUF_ELEMS * 2)
#define ZBUF_ELEMS (B_SZ * NF)                            // bf16 elems / buffer
#define WIN_OFF    (ZBUF_OFF + 3 * ZBUF_ELEMS * 2)
#define WIN_ELEMS  (22 * 64 * 512)                        // frag elems
#define WREC_OFF   (WIN_OFF + WIN_ELEMS * 2)
#define WREC_ELEMS (32 * 64 * 512)
#define WOUT_OFF   (WREC_OFF + WREC_ELEMS * 2)
#define WOUT_ELEMS (32 * 2 * 512)

// ---------------------------------------------------------------------------
// Prep: bf16-convert weights into per-lane WMMA B fragments, zero z_{-1},
// reset the grid-barrier ticket counter.  B-fragment layout (per 32x16 tile):
//   lane = (k_in_tile/16)*16 + n_in_tile ; element e: K = (k_in_tile/16)*16+e
//   -> each lane owns 16 bf16 = 32 contiguous bytes.
// ---------------------------------------------------------------------------
__global__ void snn_prep(const float* __restrict__ w_in,
                         const float* __restrict__ w_rec,
                         const float* __restrict__ w_out,
                         unsigned char* __restrict__ ws) {
  const unsigned tid  = blockIdx.x * blockDim.x + threadIdx.x;
  const unsigned nthr = gridDim.x * blockDim.x;

  if (tid == 0) *(unsigned*)(ws + BAR_OFF) = 0u;

  // zero z buffer #2 (acts as z_{-1})
  unsigned* z2 = (unsigned*)(ws + ZBUF_OFF + 2 * ZBUF_ELEMS * 2);
  for (unsigned i = tid; i < ZBUF_ELEMS / 2; i += nthr) z2[i] = 0u;

  __bf16* fin = (__bf16*)(ws + WIN_OFF);
  for (unsigned i = tid; i < WIN_ELEMS; i += nthr) {
    unsigned e = i & 15, lane = (i >> 4) & 31, f = i >> 9;
    unsigned nt = f % 64, kt = f / 64;
    unsigned k = kt * 32 + (lane >> 4) * 16 + e;
    unsigned n = nt * 16 + (lane & 15);
    float v = (k < NI) ? w_in[n * NI + k] : 0.0f;   // B[k][n] = w_in[n][k]
    fin[i] = (__bf16)v;
  }

  __bf16* frec = (__bf16*)(ws + WREC_OFF);
  for (unsigned i = tid; i < WREC_ELEMS; i += nthr) {
    unsigned e = i & 15, lane = (i >> 4) & 31, f = i >> 9;
    unsigned nt = f & 63, kt = f >> 6;
    unsigned k = kt * 32 + (lane >> 4) * 16 + e;
    unsigned n = nt * 16 + (lane & 15);
    frec[i] = (__bf16)w_rec[n * NF + k];
  }

  __bf16* fo = (__bf16*)(ws + WOUT_OFF);
  for (unsigned i = tid; i < WOUT_ELEMS; i += nthr) {
    unsigned e = i & 15, lane = (i >> 4) & 31, f = i >> 9;
    unsigned nt = f & 1, kt = f >> 1;
    unsigned k = kt * 32 + (lane >> 4) * 16 + e;
    unsigned n = nt * 16 + (lane & 15);
    float v = (n < NO) ? w_out[n * NF + k] : 0.0f;
    fo[i] = (__bf16)v;
  }
}

// ---------------------------------------------------------------------------
// fragment loaders
// ---------------------------------------------------------------------------
__device__ __forceinline__ v16bf load_a_frag(const __bf16* __restrict__ base,
                                             int stride, int kt, int mtile,
                                             int lane) {
  // A 16x32 bf16 layout: lanes 0-15 M=0..15 (K 0-15), lanes 16-31 same M
  // with K 16-31.  Two 16B loads per lane.
  int m = mtile * 16 + (lane & 15);
  const __bf16* p = base + m * stride + kt * 32 + ((lane >> 4) << 4);
  v8bf lo = *(const v8bf*)p;
  v8bf hi = *(const v8bf*)(p + 8);
  return __builtin_shufflevector(lo, hi, 0, 1, 2, 3, 4, 5, 6, 7,
                                         8, 9, 10, 11, 12, 13, 14, 15);
}

__device__ __forceinline__ v16bf load_b_frag(const __bf16* __restrict__ frags,
                                             int fidx, int lane) {
  return *(const v16bf*)(frags + fidx * 512 + lane * 16);  // 32B / lane
}

__device__ __forceinline__ v8f wmma_bf16(v16bf a, v16bf b, v8f c) {
  return __builtin_amdgcn_wmma_f32_16x16x32_bf16(false, a, false, b,
                                                 (short)0, c, false, false);
}

// monotonic ticket grid barrier (no reset race; prep re-zeroes per launch)
__device__ __forceinline__ void grid_barrier(unsigned* cnt, unsigned target) {
  __threadfence();
  __syncthreads();
  if (threadIdx.x == 0) {
    __hip_atomic_fetch_add(cnt, 1u, __ATOMIC_ACQ_REL, __HIP_MEMORY_SCOPE_AGENT);
    while (__hip_atomic_load(cnt, __ATOMIC_ACQUIRE, __HIP_MEMORY_SCOPE_AGENT) <
           target)
      __builtin_amdgcn_s_sleep(1);
  }
  __syncthreads();
  __threadfence();
}

// ---------------------------------------------------------------------------
// Persistent kernel: blocks 0..15 each own 64 hidden neurons, block 16 owns
// the LI readout.  One grid barrier per timestep.
// ---------------------------------------------------------------------------
__global__ void __launch_bounds__(THREADS, 1)
snn_run(const float* __restrict__ x, unsigned char* __restrict__ ws,
        float* __restrict__ out) {
  const int tid  = threadIdx.x;
  const int lane = tid & 31;
  const int wave = tid >> 5;
  const int blk  = blockIdx.x;

  unsigned* bar = (unsigned*)(ws + BAR_OFF);
  __bf16* xbuf  = (__bf16*)(ws + XBUF_OFF);
  __bf16* zbuf  = (__bf16*)(ws + ZBUF_OFF);
  const __bf16* fin  = (const __bf16*)(ws + WIN_OFF);
  const __bf16* frec = (const __bf16*)(ws + WREC_OFF);
  const __bf16* fout = (const __bf16*)(ws + WOUT_OFF);

  __shared__ float vo_lds[B_SZ][NO_PAD];

  const bool is_readout = (blk == NWG_HID);

  // persistent neuron state in WMMA C-fragment layout
  float vst[16], ist[16];                 // hidden: 2 tiles x 8 rows
  float vo[8], io[8];                     // readout: 1 tile
#pragma unroll
  for (int s = 0; s < 16; ++s) { vst[s] = 0.0f; ist[s] = 0.0f; }
#pragma unroll
  for (int s = 0; s < 8; ++s) { vo[s] = 0.0f; io[s] = 0.0f; }

  int mtile, nt0 = 0, nt1 = 0, r_nt = 0;
  if (!is_readout) {
    mtile = wave >> 1;                    // batch tile 0..3
    nt0 = blk * 4 + (wave & 1) * 2;       // global neuron tiles
    nt1 = nt0 + 1;
  } else {
    mtile = wave & 3;
    r_nt = wave >> 2;                     // output tile 0..1
  }

  const unsigned gtid = blk * THREADS + tid;
  const unsigned nthr = NWG_TOT * THREADS;

  for (int t = 0; t < T_STEPS; ++t) {
    __bf16* xcur = xbuf + (unsigned)(t & 1) * XBUF_ELEMS;
    __bf16* zcur = zbuf + (unsigned)(t % 3) * ZBUF_ELEMS;
    const __bf16* zprev = zbuf + (unsigned)((t + 2) % 3) * ZBUF_ELEMS;

    // ---------------- phase 1: elementwise LIF + publish z_t + stage x_t ----
    if (!is_readout) {
#pragma unroll
      for (int s = 0; s < 16; ++s) {
        float i_old = ist[s], v_old = vst[s];
        float idec = 0.8f * i_old;                    // i*(1 - dt*tau_syn_inv)
        float vdec = v_old + 0.1f * (i_old - v_old);  // v + dt*tau_mem_inv*(i-v)
        float z = (vdec > 1.0f) ? 1.0f : 0.0f;        // heaviside(v - v_th)
        vst[s] = (1.0f - z) * vdec;                   // reset on spike
        ist[s] = idec;                                // GEMM added in phase 2
        int nt = (s < 8) ? nt0 : nt1;
        int v = s & 7;
        int row = mtile * 16 + v + ((lane >> 4) << 3);
        int col = nt * 16 + (lane & 15);
        zcur[row * NF + col] = (__bf16)z;
      }
    } else {
#pragma unroll
      for (int s = 0; s < 8; ++s) {
        float io_old = io[s];
        vo[s] = vo[s] + 0.1f * (io_old - vo[s]);
        io[s] = 0.8f * io_old;                        // io_dec; GEMM in phase 2
      }
    }
    {  // all 17 blocks stage x_t -> bf16 (zero-padded to 704)
      const float* xt = x + (size_t)t * B_SZ * NI;
      for (unsigned idx = gtid; idx < B_SZ * NI_PAD; idx += nthr) {
        unsigned m = idx / NI_PAD, k = idx % NI_PAD;
        float v = (k < NI) ? xt[m * NI + k] : 0.0f;
        xcur[idx] = (__bf16)v;
      }
    }

    grid_barrier(bar, (unsigned)(t + 1) * NWG_TOT);

    // ---------------- phase 2: WMMA GEMMs -----------------------------------
    if (!is_readout) {
      v8f acc0 = {}, acc1 = {};
#pragma unroll 2
      for (int kt = 0; kt < 22; ++kt) {               // x_t @ w_in^T
        v16bf a  = load_a_frag(xcur, NI_PAD, kt, mtile, lane);
        v16bf b0 = load_b_frag(fin, kt * 64 + nt0, lane);
        v16bf b1 = load_b_frag(fin, kt * 64 + nt1, lane);
        acc0 = wmma_bf16(a, b0, acc0);
        acc1 = wmma_bf16(a, b1, acc1);
      }
#pragma unroll 2
      for (int kt = 0; kt < 32; ++kt) {               // z_{t-1} @ w_rec^T
        v16bf a  = load_a_frag(zprev, NF, kt, mtile, lane);
        v16bf b0 = load_b_frag(frec, kt * 64 + nt0, lane);
        v16bf b1 = load_b_frag(frec, kt * 64 + nt1, lane);
        acc0 = wmma_bf16(a, b0, acc0);
        acc1 = wmma_bf16(a, b1, acc1);
      }
#pragma unroll
      for (int v = 0; v < 8; ++v) {
        ist[v]     += acc0[v];
        ist[8 + v] += acc1[v];
      }
    } else {
      v8f acc = {};
#pragma unroll 2
      for (int kt = 0; kt < 32; ++kt) {               // z_t @ w_out^T
        v16bf a = load_a_frag(zcur, NF, kt, mtile, lane);
        v16bf b = load_b_frag(fout, kt * 2 + r_nt, lane);
        acc = wmma_bf16(a, b, acc);
      }
#pragma unroll
      for (int v = 0; v < 8; ++v) io[v] += acc[v];
    }
  }

  // ---------------- final: log_softmax(vo) ---------------------------------
  if (is_readout) {
#pragma unroll
    for (int v = 0; v < 8; ++v) {
      int row = mtile * 16 + v + ((lane >> 4) << 3);
      int col = r_nt * 16 + (lane & 15);
      vo_lds[row][col] = vo[v];
    }
    __syncthreads();
    for (int row = tid; row < B_SZ; row += THREADS) {
      float m = -1e30f;
      for (int c = 0; c < NO; ++c) m = fmaxf(m, vo_lds[row][c]);
      float s = 0.0f;
      for (int c = 0; c < NO; ++c) s += __expf(vo_lds[row][c] - m);
      float lse = m + __logf(s);
      for (int c = 0; c < NO; ++c) out[row * NO + c] = vo_lds[row][c] - lse;
    }
  }
}

// ---------------------------------------------------------------------------
extern "C" void kernel_launch(void* const* d_in, const int* in_sizes, int n_in,
                              void* d_out, int out_size, void* d_ws,
                              size_t ws_size, hipStream_t stream) {
  const float* x     = (const float*)d_in[0];
  const float* w_in  = (const float*)d_in[1];
  const float* w_rec = (const float*)d_in[2];
  const float* w_out = (const float*)d_in[3];
  unsigned char* ws  = (unsigned char*)d_ws;
  float* out         = (float*)d_out;

  snn_prep<<<512, THREADS, 0, stream>>>(w_in, w_rec, w_out, ws);
  snn_run<<<NWG_TOT, THREADS, 0, stream>>>(x, ws, out);
}
